// MultiHeadAttention_18468359373181
// MI455X (gfx1250) — matrix operand order repaired and checked by reading, not compiled, not run on hardware
//
#include <hip/hip_runtime.h>
#include <hip/hip_bf16.h>

typedef __bf16 bf16;
typedef __attribute__((ext_vector_type(16))) __bf16 v16bf;
typedef __attribute__((ext_vector_type(8)))  __bf16 v8bf;
typedef __attribute__((ext_vector_type(8)))  float  v8f;
typedef __attribute__((ext_vector_type(4)))  float  v4f;

#define B_   2
#define S_   2048
#define C_   768
#define NH_  12
#define HD_  64
#define QKVN (3 * C_)   // 2304

__device__ __forceinline__ v16bf cat16(v8bf lo, v8bf hi) {
    return __builtin_shufflevector(lo, hi, 0,1,2,3,4,5,6,7,8,9,10,11,12,13,14,15);
}
__device__ __forceinline__ v8f wmma_bf16(v16bf a, v16bf b, v8f c) {
    return __builtin_amdgcn_wmma_f32_16x16x32_bf16(false, a, false, b, (short)0, c, false, false);
}

// ---------------------------------------------------------------------------
// fp32 -> bf16 conversion, 8 elements/thread (2x b128 in, 1x b128 out)
// ---------------------------------------------------------------------------
__global__ void cvt_f32_bf16(const float* __restrict__ in, bf16* __restrict__ out, int n8) {
    int i = blockIdx.x * blockDim.x + threadIdx.x;
    if (i >= n8) return;
    const v4f* p = (const v4f*)(in + (size_t)i * 8);
    v4f x0 = p[0], x1 = p[1];
    v8bf o;
    o[0] = (bf16)x0[0]; o[1] = (bf16)x0[1]; o[2] = (bf16)x0[2]; o[3] = (bf16)x0[3];
    o[4] = (bf16)x1[0]; o[5] = (bf16)x1[1]; o[6] = (bf16)x1[2]; o[7] = (bf16)x1[3];
    *(v8bf*)(out + (size_t)i * 8) = o;
}

// ---------------------------------------------------------------------------
// Block-tiled WMMA GEMM:  out[M,N] = A[M,K] * W[N,K]^T + bias[N]
// Block = 128 threads (4 waves); block tile 128M x 64N; wave tile 32M x 64N.
// 64x64 W tiles staged into LDS via global_load_async_to_lds_b128 (ASYNCcnt),
// double-buffered; each thread issues 4 chained async b128 copies per buffer
// (IOFFSET applies to both LDS and global address).
// SPLITV: columns >= 1536 (V of the QKV projection) scatter transposed into
// vT[b][h][d][t] so the attention kernel gets contiguous V fragments.
// ---------------------------------------------------------------------------
template<int N, int K, bool F32OUT, bool SPLITV>
__global__ __launch_bounds__(128) void gemm_bf16_wmma(
    const bf16* __restrict__ A, const bf16* __restrict__ W,
    const float* __restrict__ bias, void* __restrict__ out,
    bf16* __restrict__ vT)
{
    __shared__ bf16 Wlds[2][64 * 64];       // 8KB per buffer

    constexpr int nblk  = N >> 6;
    constexpr int steps = K >> 6;           // 64-wide k-steps

    int bm   = blockIdx.x / nblk;
    int bn   = blockIdx.x % nblk;
    int tid  = threadIdx.x;
    int wv   = tid >> 5;
    int lane = tid & 31;
    int half = lane >> 4;
    int l16  = lane & 15;

    int n0 = bn << 6;
    int m0 = (bm << 7) + (wv << 5);         // this wave's 32 rows

    // staging map: thread copies 64B of row (tid>>1), halves [(tid&1)*32 ..+31]
    int r0 = tid >> 1;
    int c0 = (tid & 1) * 32;
    const bf16* wsrc   = W + (size_t)(n0 + r0) * K + c0;
    unsigned    lbase  = (unsigned)(uintptr_t)&Wlds[0][r0 * 64 + c0];

    const bf16* arow0 = A + (size_t)(m0 + l16) * K;
    const bf16* arow1 = A + (size_t)(m0 + 16 + l16) * K;

    v8f acc[2][4] = {};

    auto issue = [&](int buf, int k0) {
        const bf16* g = wsrc + k0;
        unsigned l = lbase + buf * 8192;
        asm volatile(
            "global_load_async_to_lds_b128 %0, %1, off\n\t"
            "global_load_async_to_lds_b128 %0, %1, off offset:16\n\t"
            "global_load_async_to_lds_b128 %0, %1, off offset:32\n\t"
            "global_load_async_to_lds_b128 %0, %1, off offset:48"
            :: "v"(l), "v"(g) : "memory");
    };

    auto compute = [&](int buf, int k0) {
        #pragma unroll
        for (int kk = 0; kk < 2; ++kk) {
            const bf16* ap0 = arow0 + k0 + kk * 32 + half * 16;
            const bf16* ap1 = arow1 + k0 + kk * 32 + half * 16;
            v16bf a0 = cat16(*(const v8bf*)ap0, *(const v8bf*)(ap0 + 8));
            v16bf a1 = cat16(*(const v8bf*)ap1, *(const v8bf*)(ap1 + 8));
            #pragma unroll
            for (int j = 0; j < 4; ++j) {
                const bf16* lp = &Wlds[buf][(j * 16 + l16) * 64 + kk * 32 + half * 16];
                v16bf bm_ = cat16(*(const v8bf*)lp, *(const v8bf*)(lp + 8));
                acc[0][j] = wmma_bf16(a0, bm_, acc[0][j]);
                acc[1][j] = wmma_bf16(a1, bm_, acc[1][j]);
            }
        }
    };

    issue(0, 0);
    for (int i = 0; i < steps - 1; ++i) {
        int buf = i & 1;
        issue(buf ^ 1, (i + 1) << 6);
        asm volatile("s_wait_asynccnt 0x4" ::: "memory");  // current buffer's 4 writes done
        __syncthreads();
        compute(buf, i << 6);
        __syncthreads();                                   // reads done before overwrite
    }
    asm volatile("s_wait_asynccnt 0x0" ::: "memory");
    __syncthreads();
    compute((steps - 1) & 1, (steps - 1) << 6);

    #pragma unroll
    for (int t = 0; t < 2; ++t) {
        #pragma unroll
        for (int j = 0; j < 4; ++j) {
            int col = n0 + j * 16 + l16;
            float bv = bias[col];
            #pragma unroll
            for (int v = 0; v < 8; ++v) {
                int row = m0 + t * 16 + v + half * 8;
                float r = acc[t][j][v] + bv;
                if constexpr (F32OUT) {
                    ((float*)out)[(size_t)row * N + col] = r;
                } else if constexpr (SPLITV) {
                    if (col >= 2 * C_) {   // block-uniform (1536 % 64 == 0)
                        int d  = (col - 2 * C_) & (HD_ - 1);
                        int hh = (col - 2 * C_) >> 6;
                        int bb = row >> 11;            // / S_
                        int ss = row & (S_ - 1);
                        vT[(((size_t)bb * NH_ + hh) * HD_ + d) * S_ + ss] = (bf16)r;
                    } else {
                        ((bf16*)out)[(size_t)row * N + col] = (bf16)r;
                    }
                } else {
                    ((bf16*)out)[(size_t)row * N + col] = (bf16)r;
                }
            }
        }
    }
}

// ---------------------------------------------------------------------------
// Flash attention: 1 wave per (b, h, 16-row Q tile). Online softmax over t in
// chunks of 64 (4 column-tiles): one row-stat update, one LDS P restage and
// one barrier pair per 64 columns. Q,K from qkv [B*S,2304]; V from transposed
// vT[b][h][d][t] so K and V fragments are contiguous 32B loads.
// ---------------------------------------------------------------------------
__global__ __launch_bounds__(32) void flash_attn_wmma(
    const bf16* __restrict__ qkv, const bf16* __restrict__ vT,
    const unsigned char* __restrict__ mask, bf16* __restrict__ attn_out)
{
    __shared__ bf16 Plds[16 * 64];          // 2KB P staging

    int wid = blockIdx.x;
    int mt  = wid & (S_ / 16 - 1);
    int h   = (wid / (S_ / 16)) % NH_;
    int b   = wid / ((S_ / 16) * NH_);
    int lane = threadIdx.x & 31;
    int half = lane >> 4;
    int l16  = lane & 15;

    const size_t rowB = (size_t)b * S_;
    const bf16* qbase = qkv + rowB * QKVN + h * HD_;
    const bf16* kbase = qbase + C_;
    const bf16* vtb   = vT + ((size_t)(b * NH_ + h) * HD_) * S_;

    // Q fragments: 16x64 -> two 16x32 A fragments
    v16bf qf[2];
    {
        const bf16* qr = qbase + (size_t)(mt * 16 + l16) * QKVN;
        #pragma unroll
        for (int kk = 0; kk < 2; ++kk) {
            const bf16* p = qr + kk * 32 + half * 16;
            qf[kk] = cat16(*(const v8bf*)p, *(const v8bf*)(p + 8));
        }
    }

    float mrun[8], lrun[8];
    v8f o[4] = {};
    #pragma unroll
    for (int v = 0; v < 8; ++v) { mrun[v] = -3.0e38f; lrun[v] = 0.0f; }

    const float scale = 0.125f;  // 1/sqrt(HD)

    for (int t0 = 0; t0 < S_; t0 += 64) {
        // ---- scores: four 16x16 tiles over 64 t-columns ----
        v8f sc[4] = {};
        #pragma unroll
        for (int kk = 0; kk < 2; ++kk) {
            #pragma unroll
            for (int c = 0; c < 4; ++c) {
                const bf16* kp = kbase + (size_t)(t0 + c * 16 + l16) * QKVN + kk * 32 + half * 16;
                sc[c] = wmma_bf16(qf[kk], *(const v16bf*)kp, sc[c]);
            }
        }

        bool mk[4];
        #pragma unroll
        for (int c = 0; c < 4; ++c) mk[c] = mask[rowB + t0 + c * 16 + l16] != 0;

        float f[4][8];
        #pragma unroll
        for (int c = 0; c < 4; ++c)
            #pragma unroll
            for (int v = 0; v < 8; ++v)
                f[c][v] = mk[c] ? -3.0e38f : sc[c][v] * scale;

        // ---- online softmax per row (row v+half*8 lives in one half-wave) ----
        #pragma unroll
        for (int v = 0; v < 8; ++v) {
            float mloc = fmaxf(fmaxf(f[0][v], f[1][v]), fmaxf(f[2][v], f[3][v]));
            #pragma unroll
            for (int off = 8; off > 0; off >>= 1)
                mloc = fmaxf(mloc, __shfl_xor(mloc, off, 32));
            float mnew  = fmaxf(mrun[v], mloc);
            float alpha = __expf(mrun[v] - mnew);
            float e[4], es = 0.0f;
            #pragma unroll
            for (int c = 0; c < 4; ++c) { e[c] = __expf(f[c][v] - mnew); es += e[c]; }
            #pragma unroll
            for (int off = 8; off > 0; off >>= 1)
                es += __shfl_xor(es, off, 32);
            lrun[v] = lrun[v] * alpha + es;
            mrun[v] = mnew;
            #pragma unroll
            for (int j = 0; j < 4; ++j) o[j][v] *= alpha;
            #pragma unroll
            for (int c = 0; c < 4; ++c)
                Plds[(v + half * 8) * 64 + c * 16 + l16] = (bf16)e[c];
        }
        __syncthreads();
        // read P back as two 16x32 A fragments
        v16bf pa[2];
        #pragma unroll
        for (int kk = 0; kk < 2; ++kk) {
            const bf16* pp = &Plds[l16 * 64 + kk * 32 + half * 16];
            pa[kk] = cat16(*(const v8bf*)pp, *(const v8bf*)(pp + 8));
        }
        __syncthreads();

        // ---- PV: 4 d-tiles x 2 k-subtiles; V fragments are 32B loads ----
        #pragma unroll
        for (int kk = 0; kk < 2; ++kk) {
            #pragma unroll
            for (int j = 0; j < 4; ++j) {
                const bf16* vp = vtb + (size_t)(j * 16 + l16) * S_ + t0 + kk * 32 + half * 16;
                o[j] = wmma_bf16(pa[kk], *(const v16bf*)vp, o[j]);
            }
        }
    }

    // ---- epilogue: normalize and store bf16 (b, s, h, d) ----
    bf16* outp = attn_out + (rowB + mt * 16) * C_ + h * HD_;
    #pragma unroll
    for (int j = 0; j < 4; ++j) {
        #pragma unroll
        for (int v = 0; v < 8; ++v) {
            int row = v + half * 8;
            outp[(size_t)row * C_ + j * 16 + l16] = (bf16)(o[j][v] / lrun[v]);
        }
    }
}

// ---------------------------------------------------------------------------
extern "C" void kernel_launch(void* const* d_in, const int* in_sizes, int n_in,
                              void* d_out, int out_size, void* d_ws, size_t ws_size,
                              hipStream_t stream)
{
    (void)in_sizes; (void)n_in; (void)out_size; (void)ws_size;

    const float*         x      = (const float*)d_in[0];
    const unsigned char* mask   = (const unsigned char*)d_in[1];
    const float*         Wqkv_w = (const float*)d_in[2];
    const float*         Wqkv_b = (const float*)d_in[3];
    const float*         Wo_w   = (const float*)d_in[4];
    const float*         Wo_b   = (const float*)d_in[5];
    float*               out    = (float*)d_out;

    const size_t nX    = (size_t)B_ * S_ * C_;      // 3,145,728
    const size_t nWqkv = (size_t)QKVN * C_;         // 1,769,472
    const size_t nWo   = (size_t)C_ * C_;           //   589,824
    const size_t nQKV  = (size_t)B_ * S_ * QKVN;    // 9,437,184

    char* ws = (char*)d_ws;
    size_t off = 0;
    auto take = [&](size_t bytes) { char* p = ws + off; off += (bytes + 255) & ~(size_t)255; return p; };
    bf16* x_bf    = (bf16*)take(nX * 2);
    bf16* wqkv_bf = (bf16*)take(nWqkv * 2);
    bf16* wo_bf   = (bf16*)take(nWo * 2);
    bf16* qkv_bf  = (bf16*)take(nQKV * 2);
    bf16* vT_bf   = (bf16*)take(nX * 2);            // [B][NH][HD][S]
    bf16* attn_bf = (bf16*)take(nX * 2);

    // 1) downconvert inputs to bf16 (8 elems/thread)
    cvt_f32_bf16<<<(int)((nX / 8    + 255) / 256), 256, 0, stream>>>(x,      x_bf,    (int)(nX / 8));
    cvt_f32_bf16<<<(int)((nWqkv / 8 + 255) / 256), 256, 0, stream>>>(Wqkv_w, wqkv_bf, (int)(nWqkv / 8));
    cvt_f32_bf16<<<(int)((nWo / 8   + 255) / 256), 256, 0, stream>>>(Wo_w,   wo_bf,   (int)(nWo / 8));

    // 2) QKV projection: q,k into qkv_bf, v transposed into vT
    {
        int blocks = ((B_ * S_) / 128) * (QKVN / 64);   // 32 * 36 = 1152
        gemm_bf16_wmma<QKVN, C_, false, true>
            <<<blocks, 128, 0, stream>>>(x_bf, wqkv_bf, Wqkv_b, qkv_bf, vT_bf);
    }

    // 3) attention: 1 wave per (b, h, 16-row tile)
    flash_attn_wmma<<<B_ * NH_ * (S_ / 16), 32, 0, stream>>>(qkv_bf, vT_bf, mask, attn_bf);

    // 4) output projection -> fp32 d_out
    {
        int blocks = ((B_ * S_) / 128) * (C_ / 64);     // 32 * 12 = 384
        gemm_bf16_wmma<C_, C_, true, false>
            <<<blocks, 128, 0, stream>>>(attn_bf, wo_bf, Wo_b, out, nullptr);
    }
}
